// MetaLayer_53798760350348
// MI455X (gfx1250) — matrix operand order repaired and checked by reading, not compiled, not run on hardware
//
#include <hip/hip_runtime.h>
#include <hip/hip_bf16.h>

typedef __attribute__((ext_vector_type(16))) __bf16 v16bf;
typedef __attribute__((ext_vector_type(8)))  __bf16 v8bf;
typedef __attribute__((ext_vector_type(8)))  float  v8f;

#define N_NODES 65536
#define N_EDGES 262144
#define N_FACES 32768
#define N_GRAPH 128
#define DIM     128

// ------------------------------------------------------------------
// Pack a [K x 128] f32 weight matrix into bf16 WMMA B-fragment layout.
// Fragment (kt,n) covers K rows [kt*32, kt*32+32) and cols [n*16, n*16+16).
// Within a fragment, lane l supplies 16 K values of column n*16+(l&15),
// matching the A layout: lanes 0-15 hold K {0..7,16..23}, lanes 16-31 hold
// K {8..15,24..31} (offset kt*32), 2 halfs per VGPR.
// Packed element index = ((kt*8 + n)*32 + lane)*16 + j.
// ------------------------------------------------------------------
__global__ void pack_w_kernel(const float* __restrict__ W, __bf16* __restrict__ out, int K) {
    int idx = blockIdx.x * blockDim.x + threadIdx.x;
    if (idx >= K * DIM) return;
    int j    = idx & 15;
    int lane = (idx >> 4) & 31;
    int frag = idx >> 9;
    int n    = frag & 7;       // 128/16 = 8 column tiles
    int kt   = frag >> 3;
    int k    = kt * 32 + ((lane & 16) ? 8 : 0) + (j < 8 ? j : j + 8);
    int col  = n * 16 + (lane & 15);
    out[idx] = (__bf16)W[(size_t)k * DIM + col];
}

__global__ void zero_kernel(float* __restrict__ p, long long n) {
    long long i = (long long)blockIdx.x * blockDim.x + threadIdx.x;
    long long s = (long long)gridDim.x * blockDim.x;
    for (; i < n; i += s) p[i] = 0.f;
}

// ------------------------------------------------------------------
// Two-layer MLP over a 32-row tile staged in LDS (bf16), 64 threads.
// Wave w computes rows [w*16, w*16+16).  Layer1: Kw -> 128 (+bias, ReLU),
// restaged to LDS bf16.  Layer2: 128 -> 128 (+bias), returned in acc2.
// A-fragment layout (16-bit A 16x32): lane<16 holds K {0..7,16..23},
// lane>=16 holds K {8..15,24..31}; M = lane&15.
// ------------------------------------------------------------------
template<int KT1>
__device__ __forceinline__ void mlp2_tile(const __bf16* __restrict__ ldsA, int Kw,
                                          __bf16* __restrict__ ldsH,
                                          const __bf16* __restrict__ W1p, const float* __restrict__ b1,
                                          const __bf16* __restrict__ W2p, const float* __restrict__ b2,
                                          v8f acc2[8]) {
    const int tid  = threadIdx.x;
    const int wave = tid >> 5;
    const int lane = tid & 31;
    const int mrow = lane & 15;
    const int ka   = (lane & 16) ? 8 : 0;   // element offset into 32-wide K slice

    v8f acc[8];
#pragma unroll
    for (int n = 0; n < 8; ++n) {
#pragma unroll
        for (int i = 0; i < 8; ++i) acc[n][i] = 0.f;
    }

    for (int kt = 0; kt < KT1; ++kt) {
        const __bf16* ap = ldsA + (size_t)(wave * 16 + mrow) * Kw + kt * 32 + ka;
        v8bf lo = *(const v8bf*)(ap);
        v8bf hi = *(const v8bf*)(ap + 16);
        v16bf a;
#pragma unroll
        for (int i = 0; i < 8; ++i) { a[i] = lo[i]; a[i + 8] = hi[i]; }
        const v16bf* bf = (const v16bf*)(W1p + (size_t)kt * 8 * 512);
#pragma unroll
        for (int n = 0; n < 8; ++n) {
            v16bf b = bf[(size_t)n * 32 + lane];
            acc[n] = __builtin_amdgcn_wmma_f32_16x16x32_bf16(
                false, a, false, b, (short)0, acc[n], false, false);
        }
    }

    // bias + ReLU, restage hidden as bf16 (C/D layout: VGPR r -> M = r + (lane>=16 ? 8:0))
    const int rbase = wave * 16 + ((lane & 16) ? 8 : 0);
#pragma unroll
    for (int n = 0; n < 8; ++n) {
        const int col = n * 16 + mrow;
        const float bias = b1[col];
#pragma unroll
        for (int r = 0; r < 8; ++r) {
            float h = acc[n][r] + bias;
            ldsH[(size_t)(rbase + r) * DIM + col] = (__bf16)(h > 0.f ? h : 0.f);
        }
    }
    __syncthreads();

#pragma unroll
    for (int n = 0; n < 8; ++n) {
#pragma unroll
        for (int i = 0; i < 8; ++i) acc2[n][i] = 0.f;
    }
    for (int kt = 0; kt < 4; ++kt) {
        const __bf16* ap = ldsH + (size_t)(wave * 16 + mrow) * DIM + kt * 32 + ka;
        v8bf lo = *(const v8bf*)(ap);
        v8bf hi = *(const v8bf*)(ap + 16);
        v16bf a;
#pragma unroll
        for (int i = 0; i < 8; ++i) { a[i] = lo[i]; a[i + 8] = hi[i]; }
        const v16bf* bf = (const v16bf*)(W2p + (size_t)kt * 8 * 512);
#pragma unroll
        for (int n = 0; n < 8; ++n) {
            v16bf b = bf[(size_t)n * 32 + lane];
            acc2[n] = __builtin_amdgcn_wmma_f32_16x16x32_bf16(
                false, a, false, b, (short)0, acc2[n], false, false);
        }
    }
#pragma unroll
    for (int n = 0; n < 8; ++n) {
        const float bias = b2[n * 16 + mrow];
#pragma unroll
        for (int r = 0; r < 8; ++r) acc2[n][r] += bias;
    }
}

// ------------------------------------------------------------------
// Edge update: e_in = [edge_attr | x[row] | x[col] | u[b] | face[f0] | face[f1]]
// ------------------------------------------------------------------
__global__ __launch_bounds__(64) void edge_kernel(
    const float* __restrict__ x, const float* __restrict__ ea,
    const float* __restrict__ u, const float* __restrict__ face,
    const int* __restrict__ erow, const int* __restrict__ ecol,
    const int* __restrict__ f0i, const int* __restrict__ f1i,
    const __bf16* __restrict__ W1p, const float* __restrict__ b1,
    const __bf16* __restrict__ W2p, const float* __restrict__ b2,
    float* __restrict__ out_e, float* __restrict__ sent, float* __restrict__ recv,
    float* __restrict__ fsent, float* __restrict__ frecv, float* __restrict__ edge_aggr)
{
    __shared__ __align__(16) __bf16 ldsA[32 * 768];
    __shared__ __align__(16) __bf16 ldsH[32 * DIM];
    __shared__ int sR[32], sC[32], sF0[32], sF1[32];
    __shared__ float red[DIM];

    const int tid = threadIdx.x;
    const int e0  = blockIdx.x * 32;
    if (tid < 32) {
        int e = e0 + tid;
        sR[tid]  = erow[e];  sC[tid]  = ecol[e];
        sF0[tid] = f0i[e];   sF1[tid] = f1i[e];
    }
    red[tid] = 0.f; red[tid + 64] = 0.f;
    __syncthreads();

    const int batch = e0 >> 11;          // E / B = 2048 edges per graph
    for (int i = tid; i < 32 * 768; i += 64) {
        int lr  = i / 768;
        int cg  = i - lr * 768;
        int seg = cg >> 7, c = cg & 127;
        float v;
        switch (seg) {
            case 0:  v = ea  [(size_t)(e0 + lr) * DIM + c]; break;
            case 1:  v = x   [(size_t)sR [lr]   * DIM + c]; break;
            case 2:  v = x   [(size_t)sC [lr]   * DIM + c]; break;
            case 3:  v = u   [(size_t)batch     * DIM + c]; break;
            case 4:  v = face[(size_t)sF0[lr]   * DIM + c]; break;
            default: v = face[(size_t)sF1[lr]   * DIM + c]; break;
        }
        ldsA[i] = (__bf16)v;
    }
    __syncthreads();

    v8f acc2[8];
    mlp2_tile<24>(ldsA, 768, ldsH, W1p, b1, W2p, b2, acc2);

    const int lane = tid & 31;
    const int wave = tid >> 5;
    const int mrow = lane & 15;
    const int rbase = wave * 16 + ((lane & 16) ? 8 : 0);
#pragma unroll
    for (int n = 0; n < 8; ++n) {
        const int col = n * 16 + mrow;
#pragma unroll
        for (int r = 0; r < 8; ++r) {
            const int lr = rbase + r;
            const float v = acc2[n][r];
            out_e[(size_t)(e0 + lr) * DIM + col] = v;
            atomicAdd(sent  + (size_t)sR [lr] * DIM + col, v);
            atomicAdd(recv  + (size_t)sC [lr] * DIM + col, v);
            atomicAdd(fsent + (size_t)sF0[lr] * DIM + col, v);
            atomicAdd(frecv + (size_t)sF1[lr] * DIM + col, v);
            atomicAdd(&red[col], v);
        }
    }
    __syncthreads();
    atomicAdd(edge_aggr + (size_t)batch * DIM + tid,      red[tid]);
    atomicAdd(edge_aggr + (size_t)batch * DIM + tid + 64, red[tid + 64]);
}

// ------------------------------------------------------------------
// Node update: n_in = [x | sent | recv | u[b]]
// ------------------------------------------------------------------
__global__ __launch_bounds__(64) void node_kernel(
    const float* __restrict__ x, const float* __restrict__ sent,
    const float* __restrict__ recv, const float* __restrict__ u,
    const __bf16* __restrict__ W1p, const float* __restrict__ b1,
    const __bf16* __restrict__ W2p, const float* __restrict__ b2,
    float* __restrict__ out_x, float* __restrict__ node_aggr)
{
    __shared__ __align__(16) __bf16 ldsA[32 * 512];
    __shared__ __align__(16) __bf16 ldsH[32 * DIM];
    __shared__ float red[DIM];

    const int tid = threadIdx.x;
    const int n0  = blockIdx.x * 32;
    red[tid] = 0.f; red[tid + 64] = 0.f;
    const int batch = n0 >> 9;           // N / B = 512
    for (int i = tid; i < 32 * 512; i += 64) {
        int lr  = i >> 9;
        int cg  = i & 511;
        int seg = cg >> 7, c = cg & 127;
        int node = n0 + lr;
        float v;
        switch (seg) {
            case 0:  v = x   [(size_t)node  * DIM + c]; break;
            case 1:  v = sent[(size_t)node  * DIM + c]; break;
            case 2:  v = recv[(size_t)node  * DIM + c]; break;
            default: v = u   [(size_t)batch * DIM + c]; break;
        }
        ldsA[i] = (__bf16)v;
    }
    __syncthreads();

    v8f acc2[8];
    mlp2_tile<16>(ldsA, 512, ldsH, W1p, b1, W2p, b2, acc2);

    const int lane = tid & 31;
    const int wave = tid >> 5;
    const int mrow = lane & 15;
    const int rbase = wave * 16 + ((lane & 16) ? 8 : 0);
#pragma unroll
    for (int n = 0; n < 8; ++n) {
        const int col = n * 16 + mrow;
#pragma unroll
        for (int r = 0; r < 8; ++r) {
            const float v = acc2[n][r];
            out_x[(size_t)(n0 + rbase + r) * DIM + col] = v;
            atomicAdd(&red[col], v);
        }
    }
    __syncthreads();
    atomicAdd(node_aggr + (size_t)batch * DIM + tid,      red[tid]);
    atomicAdd(node_aggr + (size_t)batch * DIM + tid + 64, red[tid + 64]);
}

// ------------------------------------------------------------------
// Face update: f_in = [face | f_sent | f_recv | u[b]], then mask, aggregate.
// ------------------------------------------------------------------
__global__ __launch_bounds__(64) void face_kernel(
    const float* __restrict__ face, const float* __restrict__ fsent,
    const float* __restrict__ frecv, const float* __restrict__ u,
    const unsigned char* __restrict__ mask,
    const __bf16* __restrict__ W1p, const float* __restrict__ b1,
    const __bf16* __restrict__ W2p, const float* __restrict__ b2,
    float* __restrict__ out_f, float* __restrict__ face_aggr)
{
    __shared__ __align__(16) __bf16 ldsA[32 * 512];
    __shared__ __align__(16) __bf16 ldsH[32 * DIM];
    __shared__ float red[DIM];

    const int tid = threadIdx.x;
    const int f0  = blockIdx.x * 32;
    red[tid] = 0.f; red[tid + 64] = 0.f;
    const int batch = f0 >> 8;           // F / B = 256
    for (int i = tid; i < 32 * 512; i += 64) {
        int lr  = i >> 9;
        int cg  = i & 511;
        int seg = cg >> 7, c = cg & 127;
        int f = f0 + lr;
        float v;
        switch (seg) {
            case 0:  v = face [(size_t)f     * DIM + c]; break;
            case 1:  v = fsent[(size_t)f     * DIM + c]; break;
            case 2:  v = frecv[(size_t)f     * DIM + c]; break;
            default: v = u    [(size_t)batch * DIM + c]; break;
        }
        ldsA[i] = (__bf16)v;
    }
    __syncthreads();

    v8f acc2[8];
    mlp2_tile<16>(ldsA, 512, ldsH, W1p, b1, W2p, b2, acc2);

    const int lane = tid & 31;
    const int wave = tid >> 5;
    const int mrow = lane & 15;
    const int rbase = wave * 16 + ((lane & 16) ? 8 : 0);
#pragma unroll
    for (int n = 0; n < 8; ++n) {
        const int col = n * 16 + mrow;
#pragma unroll
        for (int r = 0; r < 8; ++r) {
            const int f = f0 + rbase + r;
            float v = acc2[n][r];
            if (mask[f]) v = 0.f;
            out_f[(size_t)f * DIM + col] = v;
            atomicAdd(&red[col], v);
        }
    }
    __syncthreads();
    atomicAdd(face_aggr + (size_t)batch * DIM + tid,      red[tid]);
    atomicAdd(face_aggr + (size_t)batch * DIM + tid + 64, red[tid + 64]);
}

// ------------------------------------------------------------------
// Global update: g_in = [u | node_aggr | edge_aggr | face_aggr]
// ------------------------------------------------------------------
__global__ __launch_bounds__(64) void global_kernel(
    const float* __restrict__ u, const float* __restrict__ na,
    const float* __restrict__ eag, const float* __restrict__ fa,
    const __bf16* __restrict__ W1p, const float* __restrict__ b1,
    const __bf16* __restrict__ W2p, const float* __restrict__ b2,
    float* __restrict__ out_u)
{
    __shared__ __align__(16) __bf16 ldsA[32 * 512];
    __shared__ __align__(16) __bf16 ldsH[32 * DIM];

    const int tid = threadIdx.x;
    const int g0  = blockIdx.x * 32;
    for (int i = tid; i < 32 * 512; i += 64) {
        int lr  = i >> 9;
        int cg  = i & 511;
        int seg = cg >> 7, c = cg & 127;
        int g = g0 + lr;
        float v;
        switch (seg) {
            case 0:  v = u  [(size_t)g * DIM + c]; break;
            case 1:  v = na [(size_t)g * DIM + c]; break;
            case 2:  v = eag[(size_t)g * DIM + c]; break;
            default: v = fa [(size_t)g * DIM + c]; break;
        }
        ldsA[i] = (__bf16)v;
    }
    __syncthreads();

    v8f acc2[8];
    mlp2_tile<16>(ldsA, 512, ldsH, W1p, b1, W2p, b2, acc2);

    const int lane = tid & 31;
    const int wave = tid >> 5;
    const int mrow = lane & 15;
    const int rbase = wave * 16 + ((lane & 16) ? 8 : 0);
#pragma unroll
    for (int n = 0; n < 8; ++n) {
        const int col = n * 16 + mrow;
#pragma unroll
        for (int r = 0; r < 8; ++r)
            out_u[(size_t)(g0 + rbase + r) * DIM + col] = acc2[n][r];
    }
}

// ------------------------------------------------------------------
extern "C" void kernel_launch(void* const* d_in, const int* in_sizes, int n_in,
                              void* d_out, int out_size, void* d_ws, size_t ws_size,
                              hipStream_t stream) {
    const float* x    = (const float*)d_in[0];
    const float* ea   = (const float*)d_in[1];
    const float* u    = (const float*)d_in[2];
    const float* face = (const float*)d_in[3];
    const int*   ei   = (const int*)d_in[4];
    const int*   fi   = (const int*)d_in[5];
    const unsigned char* mask = (const unsigned char*)d_in[12];
    const float* eW1 = (const float*)d_in[13]; const float* eb1 = (const float*)d_in[14];
    const float* eW2 = (const float*)d_in[15]; const float* eb2 = (const float*)d_in[16];
    const float* nW1 = (const float*)d_in[17]; const float* nb1 = (const float*)d_in[18];
    const float* nW2 = (const float*)d_in[19]; const float* nb2 = (const float*)d_in[20];
    const float* fW1 = (const float*)d_in[21]; const float* fb1 = (const float*)d_in[22];
    const float* fW2 = (const float*)d_in[23]; const float* fb2 = (const float*)d_in[24];
    const float* gW1 = (const float*)d_in[25]; const float* gb1 = (const float*)d_in[26];
    const float* gW2 = (const float*)d_in[27]; const float* gb2 = (const float*)d_in[28];

    float* out = (float*)d_out;
    const size_t OUT_X = 0;
    const size_t OUT_E = (size_t)N_NODES * DIM;
    const size_t OUT_U = OUT_E + (size_t)N_EDGES * DIM;
    const size_t OUT_F = OUT_U + (size_t)N_GRAPH * DIM;

    char* ws = (char*)d_ws;
    size_t off = 0;
    auto take = [&](size_t bytes) { size_t o = off; off = (off + bytes + 255) & ~(size_t)255; return o; };
    __bf16* wE1 = (__bf16*)(ws + take(768 * DIM * 2));
    __bf16* wE2 = (__bf16*)(ws + take(128 * DIM * 2));
    __bf16* wN1 = (__bf16*)(ws + take(512 * DIM * 2));
    __bf16* wN2 = (__bf16*)(ws + take(128 * DIM * 2));
    __bf16* wF1 = (__bf16*)(ws + take(512 * DIM * 2));
    __bf16* wF2 = (__bf16*)(ws + take(128 * DIM * 2));
    __bf16* wG1 = (__bf16*)(ws + take(512 * DIM * 2));
    __bf16* wG2 = (__bf16*)(ws + take(128 * DIM * 2));
    size_t accStart = off;
    float* sent  = (float*)(ws + take((size_t)N_NODES * DIM * 4));
    float* recv  = (float*)(ws + take((size_t)N_NODES * DIM * 4));
    float* fsent = (float*)(ws + take((size_t)N_FACES * DIM * 4));
    float* frecv = (float*)(ws + take((size_t)N_FACES * DIM * 4));
    float* nagg  = (float*)(ws + take((size_t)N_GRAPH * DIM * 4));
    float* eagg  = (float*)(ws + take((size_t)N_GRAPH * DIM * 4));
    float* fagg  = (float*)(ws + take((size_t)N_GRAPH * DIM * 4));
    const long long accFloats = (long long)(off - accStart) / 4;

    // 1) pack weights to bf16 WMMA fragment layout
    pack_w_kernel<<<(768 * DIM + 255) / 256, 256, 0, stream>>>(eW1, wE1, 768);
    pack_w_kernel<<<(128 * DIM + 255) / 256, 256, 0, stream>>>(eW2, wE2, 128);
    pack_w_kernel<<<(512 * DIM + 255) / 256, 256, 0, stream>>>(nW1, wN1, 512);
    pack_w_kernel<<<(128 * DIM + 255) / 256, 256, 0, stream>>>(nW2, wN2, 128);
    pack_w_kernel<<<(512 * DIM + 255) / 256, 256, 0, stream>>>(fW1, wF1, 512);
    pack_w_kernel<<<(128 * DIM + 255) / 256, 256, 0, stream>>>(fW2, wF2, 128);
    pack_w_kernel<<<(512 * DIM + 255) / 256, 256, 0, stream>>>(gW1, wG1, 512);
    pack_w_kernel<<<(128 * DIM + 255) / 256, 256, 0, stream>>>(gW2, wG2, 128);

    // 2) zero scatter accumulators
    zero_kernel<<<2048, 256, 0, stream>>>((float*)(ws + accStart), accFloats);

    // 3) edge MLP + scatter
    edge_kernel<<<N_EDGES / 32, 64, 0, stream>>>(
        x, ea, u, face, ei, ei + N_EDGES, fi, fi + N_EDGES,
        wE1, eb1, wE2, eb2, out + OUT_E, sent, recv, fsent, frecv, eagg);

    // 4) node MLP + aggregate
    node_kernel<<<N_NODES / 32, 64, 0, stream>>>(
        x, sent, recv, u, wN1, nb1, wN2, nb2, out + OUT_X, nagg);

    // 5) face MLP + mask + aggregate
    face_kernel<<<N_FACES / 32, 64, 0, stream>>>(
        face, fsent, frecv, u, mask, wF1, fb1, wF2, fb2, out + OUT_F, fagg);

    // 6) global MLP
    global_kernel<<<N_GRAPH / 32, 64, 0, stream>>>(
        u, nagg, eagg, fagg, wG1, gb1, wG2, gb2, out + OUT_U);
}
